// Qwen3MoeAttention_77584289235436
// MI455X (gfx1250) — matrix operand order repaired and checked by reading, not compiled, not run on hardware
//
#include <hip/hip_runtime.h>
#include <hip/hip_bf16.h>
#include <math.h>

// ---------------- problem constants ----------------
#define B_    2
#define S_    2048
#define HID_  2048
#define HQ_   32
#define HKV_  4
#define D_    128
#define M_    (B_*S_)              // 4096 tokens
#define SCALE_ 0.08838834764831845f // 128^-0.5

typedef __attribute__((ext_vector_type(16))) __bf16 v16bf;
typedef __attribute__((ext_vector_type(8)))  __bf16 v8bf;
typedef __attribute__((ext_vector_type(8)))  float  v8f;

// Build a 16-element bf16 A/B fragment from two contiguous 8-element (16B) runs.
__device__ __forceinline__ v16bf ld_frag(const __bf16* lo, const __bf16* hi) {
    v16bf r;
    v8bf a = *(const v8bf*)lo;
    v8bf b = *(const v8bf*)hi;
#pragma unroll
    for (int i = 0; i < 8; ++i) { r[i] = a[i]; r[8 + i] = b[i]; }
    return r;
}

// ---------------- elementwise converts ----------------
__global__ void __launch_bounds__(256) convert_bf16_kernel(
    const float* __restrict__ in, __bf16* __restrict__ out, size_t n) {
    size_t i = (size_t)blockIdx.x * 256 + threadIdx.x;
    if (i < n) out[i] = (__bf16)in[i];
}

// in: [K][N] f32 row-major  ->  out: [N][K] bf16 row-major
__global__ void __launch_bounds__(256) transpose_bf16_kernel(
    const float* __restrict__ in, __bf16* __restrict__ out, int K, int N) {
    size_t i = (size_t)blockIdx.x * 256 + threadIdx.x;   // over N*K
    int k = (int)(i % K);
    int n = (int)(i / K);
    out[i] = (__bf16)in[(size_t)k * N + n];
}

// vF: [B*S][HKV*D] f32  ->  Vt: [B][HKV][D][S] bf16
__global__ void __launch_bounds__(256) v_transpose_kernel(
    const float* __restrict__ vF, __bf16* __restrict__ Vt) {
    size_t i = (size_t)blockIdx.x * 256 + threadIdx.x;   // over B*HKV*D*S
    int s  = (int)(i % S_);
    size_t t = i / S_;
    int d  = (int)(t % D_);  t /= D_;
    int hk = (int)(t % HKV_);
    int b  = (int)(t / HKV_);
    Vt[i] = (__bf16)vF[((size_t)(b * S_ + s) * HKV_ + hk) * D_ + d];
}

// ---------------- WMMA GEMM:  C[M][N] f32 = A[M][K]bf16 * Bt[N][K]bf16^T ----------------
// block = 128 (4 waves). Each wave: 16 rows x 64 cols. K step 32.
__global__ void __launch_bounds__(128) gemm_bf16_kernel(
    const __bf16* __restrict__ A, const __bf16* __restrict__ Bt,
    float* __restrict__ C, int M, int N, int K) {
    const int lane  = threadIdx.x & 31;
    const int wave  = threadIdx.x >> 5;
    const int row16 = lane & 15;
    const int hh    = lane >> 4;                  // 0 | 1
    const int m0    = blockIdx.y * 64 + wave * 16;
    const int n0    = blockIdx.x * 64;

    v8f acc[4] = {};
    const __bf16* Arow = A + (size_t)(m0 + row16) * K + hh * 16;

    for (int kc = 0; kc < K; kc += 32) {
        v16bf a = ld_frag(Arow + kc, Arow + kc + 8);
#pragma unroll
        for (int nt = 0; nt < 4; ++nt) {
            const __bf16* Brow = Bt + (size_t)(n0 + nt * 16 + row16) * K + kc + hh * 16;
            v16bf b = ld_frag(Brow, Brow + 8);
            acc[nt] = __builtin_amdgcn_wmma_f32_16x16x32_bf16(
                false, a, false, b, (short)0, acc[nt], false, false);
        }
    }
#pragma unroll
    for (int nt = 0; nt < 4; ++nt)
#pragma unroll
        for (int r = 0; r < 8; ++r)
            C[(size_t)(m0 + r + 8 * hh) * N + n0 + nt * 16 + row16] = acc[nt][r];
}

// ---------------- fused RMS-norm + RoPE (+optional scale), f32 -> bf16 ----------------
// grid = M_*H, block = 32. X/out: [M][H*D]. cos/sin: [S][D].
__global__ void __launch_bounds__(32) normrope_kernel(
    const float* __restrict__ X, const float* __restrict__ cosT,
    const float* __restrict__ sinT, const float* __restrict__ w,
    __bf16* __restrict__ out, int H, float scale) {
    const int lane = threadIdx.x;
    const int h    = blockIdx.x % H;
    const int tok  = blockIdx.x / H;
    const int s    = tok & (S_ - 1);
    const float* row = X + ((size_t)tok * H + h) * D_;

    float v[4]; float ss = 0.f;
#pragma unroll
    for (int j = 0; j < 4; ++j) { v[j] = row[lane + 32 * j]; ss += v[j] * v[j]; }
#pragma unroll
    for (int off = 16; off >= 1; off >>= 1) ss += __shfl_xor(ss, off, 32);
    const float inv = rsqrtf(ss * (1.0f / 128.0f) + 1e-6f);

    float n[4];
#pragma unroll
    for (int j = 0; j < 4; ++j) n[j] = v[j] * inv * w[lane + 32 * j];

    __bf16* orow = out + ((size_t)tok * H + h) * D_;
#pragma unroll
    for (int j = 0; j < 4; ++j) {
        const int d = lane + 32 * j;
        const float rot = (j < 2) ? -n[j + 2] : n[j - 2];   // rotate_half partner d±64 = lane + 32*(j^2)
        const float val = (n[j] * cosT[(size_t)s * D_ + d] + rot * sinT[(size_t)s * D_ + d]) * scale;
        orow[d] = (__bf16)val;
    }
}

// ---------------- flash attention: 1 wave per 16-query tile ----------------
// Q: [B*S][HQ*D] bf16 (pre-scaled by SCALE). Kc: [B*S][HKV*D]. Vt: [B][HKV][D][S]. Ctx: [B*S][HQ*D].
__global__ void __launch_bounds__(32) attn_kernel(
    const __bf16* __restrict__ Q, const __bf16* __restrict__ Kc,
    const __bf16* __restrict__ Vt, __bf16* __restrict__ Ctx) {
    __shared__ __bf16 Pl[16][32];                 // P tile staging (D-layout -> A-layout)
    const int lane  = threadIdx.x;
    const int row16 = lane & 15;
    const int hh    = lane >> 4;
    const int q0    = blockIdx.x * 16;
    const int h     = blockIdx.y;
    const int b     = blockIdx.z;
    const int hk    = h >> 3;                     // HQ/HKV = 8

    // Q A-fragments: 4 K-chunks of 32 over D=128
    v16bf qa[4];
    {
        const __bf16* Qrow = Q + ((size_t)(b * S_ + q0 + row16) * HQ_ + h) * D_;
#pragma unroll
        for (int c = 0; c < 4; ++c) {
            const __bf16* p = Qrow + c * 32 + hh * 16;
            qa[c] = ld_frag(p, p + 8);
        }
    }

    v8f o[8] = {};
    float m[8], l[8];
#pragma unroll
    for (int r = 0; r < 8; ++r) { m[r] = -1e30f; l[r] = 0.f; }

    const int nblk = (q0 + 15) / 32 + 1;          // causal key-block count
    for (int kb = 0; kb < nblk; ++kb) {
        const int k0 = kb * 32;
        float pv[2][8];
#pragma unroll
        for (int t = 0; t < 2; ++t) {             // two 16-key score tiles
            v8f sc = {};
            const __bf16* Krow =
                Kc + ((size_t)(b * S_ + k0 + t * 16 + row16) * HKV_ + hk) * D_ + hh * 16;
#pragma unroll
            for (int c = 0; c < 4; ++c) {
                const __bf16* p = Krow + c * 32;
                v16bf kf = ld_frag(p, p + 8);     // B frag: contiguous 16 along d
                sc = __builtin_amdgcn_wmma_f32_16x16x32_bf16(
                    false, qa[c], false, kf, (short)0, sc, false, false);
            }
            const int key = k0 + t * 16 + row16;
#pragma unroll
            for (int r = 0; r < 8; ++r) {
                const int qr = q0 + r + 8 * hh;
                pv[t][r] = (key <= qr) ? sc[r] : -1e30f;   // causal mask
            }
        }
        // online softmax (row reductions across the 16-lane N groups)
#pragma unroll
        for (int r = 0; r < 8; ++r) {
            float mt = fmaxf(pv[0][r], pv[1][r]);
            mt = fmaxf(mt, __shfl_xor(mt, 1, 16));
            mt = fmaxf(mt, __shfl_xor(mt, 2, 16));
            mt = fmaxf(mt, __shfl_xor(mt, 4, 16));
            mt = fmaxf(mt, __shfl_xor(mt, 8, 16));
            const float mn    = fmaxf(m[r], mt);
            const float alpha = __expf(m[r] - mn);
            m[r] = mn;
            const float p0 = __expf(pv[0][r] - mn);
            const float p1 = __expf(pv[1][r] - mn);
            Pl[r + 8 * hh][row16]      = (__bf16)p0;
            Pl[r + 8 * hh][16 + row16] = (__bf16)p1;
            float rs = p0 + p1;
            rs += __shfl_xor(rs, 1, 16);
            rs += __shfl_xor(rs, 2, 16);
            rs += __shfl_xor(rs, 4, 16);
            rs += __shfl_xor(rs, 8, 16);
            l[r] = l[r] * alpha + rs;
#pragma unroll
            for (int nt = 0; nt < 8; ++nt) o[nt][r] *= alpha;
        }
        __syncthreads();                           // single-wave block: orders LDS store->load
        v16bf pa;
        {
            const __bf16* p = &Pl[row16][hh * 16];  // A frag of P (16q x 32key)
            pa = ld_frag(p, p + 8);
        }
        const __bf16* Vcol =
            Vt + ((size_t)(b * HKV_ + hk) * D_ + row16) * S_ + k0 + hh * 16;
#pragma unroll
        for (int nt = 0; nt < 8; ++nt) {           // O(16x128) += P(16x32) * V(32x128)
            const __bf16* p = Vcol + (size_t)nt * 16 * S_;
            v16bf vf = ld_frag(p, p + 8);          // B frag: contiguous 16 along key
            o[nt] = __builtin_amdgcn_wmma_f32_16x16x32_bf16(
                false, pa, false, vf, (short)0, o[nt], false, false);
        }
        __syncthreads();
    }
    // normalize + write bf16 context [b, s, h*D + d]
#pragma unroll
    for (int r = 0; r < 8; ++r) {
        const float inv = 1.0f / l[r];
        __bf16* crow = Ctx + ((size_t)(b * S_ + q0 + r + 8 * hh) * HQ_ + h) * D_;
#pragma unroll
        for (int nt = 0; nt < 8; ++nt)
            crow[nt * 16 + row16] = (__bf16)(o[nt][r] * inv);
    }
}

// ---------------- host orchestration ----------------
extern "C" void kernel_launch(void* const* d_in, const int* in_sizes, int n_in,
                              void* d_out, int out_size, void* d_ws, size_t ws_size,
                              hipStream_t stream) {
    const float* hid  = (const float*)d_in[0];
    const float* cosT = (const float*)d_in[1];
    const float* sinT = (const float*)d_in[2];
    const float* wq   = (const float*)d_in[3];
    const float* wk   = (const float*)d_in[4];
    const float* wv   = (const float*)d_in[5];
    const float* wo   = (const float*)d_in[6];
    const float* qnw  = (const float*)d_in[7];
    const float* knw  = (const float*)d_in[8];
    float* out = (float*)d_out;

    char* wsp = (char*)d_ws;
    auto alloc = [&](size_t bytes) -> char* {
        char* p = wsp;
        wsp += (bytes + 255) & ~(size_t)255;
        return p;
    };
    const size_t nHid = (size_t)M_ * HID_;          // 8388608
    const size_t nQ   = (size_t)M_ * HQ_ * D_;      // 16777216
    const size_t nKV  = (size_t)M_ * HKV_ * D_;     // 2097152

    __bf16* hB   = (__bf16*)alloc(nHid * 2);
    __bf16* wqT  = (__bf16*)alloc((size_t)HID_ * HQ_ * D_ * 2);
    __bf16* wkT  = (__bf16*)alloc((size_t)HID_ * HKV_ * D_ * 2);
    __bf16* wvT  = (__bf16*)alloc((size_t)HID_ * HKV_ * D_ * 2);
    __bf16* woT  = (__bf16*)alloc((size_t)HQ_ * D_ * HID_ * 2);
    float*  qF   = (float*) alloc(nQ * 4);
    float*  kF   = (float*) alloc(nKV * 4);
    float*  vF   = (float*) alloc(nKV * 4);
    __bf16* qB   = (__bf16*)alloc(nQ * 2);
    __bf16* kB   = (__bf16*)alloc(nKV * 2);
    __bf16* vT   = (__bf16*)alloc(nKV * 2);
    __bf16* ctxB = (__bf16*)alloc(nQ * 2);

    // 1) converts / transposes to bf16
    convert_bf16_kernel<<<(unsigned)(nHid / 256), 256, 0, stream>>>(hid, hB, nHid);
    transpose_bf16_kernel<<<(unsigned)((size_t)HID_ * HQ_ * D_ / 256), 256, 0, stream>>>(
        wq, wqT, HID_, HQ_ * D_);
    transpose_bf16_kernel<<<(unsigned)((size_t)HID_ * HKV_ * D_ / 256), 256, 0, stream>>>(
        wk, wkT, HID_, HKV_ * D_);
    transpose_bf16_kernel<<<(unsigned)((size_t)HID_ * HKV_ * D_ / 256), 256, 0, stream>>>(
        wv, wvT, HID_, HKV_ * D_);
    transpose_bf16_kernel<<<(unsigned)((size_t)HQ_ * D_ * HID_ / 256), 256, 0, stream>>>(
        wo, woT, HQ_ * D_, HID_);

    // 2) QKV projections (WMMA)
    gemm_bf16_kernel<<<dim3(HQ_ * D_ / 64, M_ / 64), 128, 0, stream>>>(
        hB, wqT, qF, M_, HQ_ * D_, HID_);
    gemm_bf16_kernel<<<dim3(HKV_ * D_ / 64, M_ / 64), 128, 0, stream>>>(
        hB, wkT, kF, M_, HKV_ * D_, HID_);
    gemm_bf16_kernel<<<dim3(HKV_ * D_ / 64, M_ / 64), 128, 0, stream>>>(
        hB, wvT, vF, M_, HKV_ * D_, HID_);

    // 3) RMS-norm + RoPE (Q pre-scaled by D^-1/2); V layout transform
    normrope_kernel<<<M_ * HQ_, 32, 0, stream>>>(qF, cosT, sinT, qnw, qB, HQ_, SCALE_);
    normrope_kernel<<<M_ * HKV_, 32, 0, stream>>>(kF, cosT, sinT, knw, kB, HKV_, 1.0f);
    v_transpose_kernel<<<(unsigned)(nKV / 256), 256, 0, stream>>>(vF, vT);

    // 4) causal flash attention (WMMA)
    attn_kernel<<<dim3(S_ / 16, HQ_, B_), 32, 0, stream>>>(qB, kB, vT, ctxB);

    // 5) output projection (WMMA) -> f32 d_out
    gemm_bf16_kernel<<<dim3(HID_ / 64, M_ / 64), 128, 0, stream>>>(
        ctxB, woT, out, M_, HID_, HQ_ * D_);

    (void)in_sizes; (void)n_in; (void)out_size; (void)ws_size;
}
